// TransMIL_11501922418817
// MI455X (gfx1250) — matrix operand order repaired and checked by reading, not compiled, not run on hardware
//
#include <hip/hip_runtime.h>
#include <math.h>

#define HEADS 8
#define DHEAD 16
#define LMK   64
#define EDIM  128
#define QKVN  384
static constexpr float SCALEF = 0.25f;  // DIM_HEAD^-0.5

typedef __attribute__((ext_vector_type(16))) _Float16 v16h;
typedef __attribute__((ext_vector_type(8)))  _Float16 v8h;
typedef __attribute__((ext_vector_type(8)))  float    v8f;

// fp32 -> f16 weight pre-conversion (run once per call; weights are tiny)
__global__ __launch_bounds__(256)
void k_f16cvt(const float* __restrict__ src, _Float16* __restrict__ dst, int n)
{
    const int i = blockIdx.x * 256 + threadIdx.x;
    if (i < n) dst[i] = (_Float16)src[i];
}

// ---------------------------------------------------------------------------
// WMMA GEMM:  C[M,N] = A[M,K] (fp32, row-major) * Wh[N,K]^T (f16)  (+bias)(+res)
// grid = (ceil(M/128), N/128), block = 256 (8 waves). Each wave: 16 rows x 128 cols.
// B tile (128 x 32 f16 = 8 KB) staged into LDS via async DMA, double buffered;
// the DMA for K+32 overlaps WMMA on K. A is streamed from global (HBM-bound
// operand), converted fp32->f16 in-register (free vs HBM time). All 8 B
// fragments are preloaded into distinct registers so the ds_loads batch under
// a single s_wait_dscnt instead of stalling each WMMA.
// ---------------------------------------------------------------------------
#define BSTRIDE 40   // halfs per LDS row (80 B): 16B-aligned, spreads banks

__global__ __launch_bounds__(256)
void k_gemm(const float* __restrict__ A, const _Float16* __restrict__ Wh,
            const float* __restrict__ bias, const float* __restrict__ res,
            float* __restrict__ C, int M, int K, int N)
{
    __shared__ _Float16 Bs[2][128 * BSTRIDE];   // 20 KB
    const int tid  = threadIdx.x;
    const int lane = tid & 31;
    const int wave = tid >> 5;
    const int r    = lane & 15;
    const int half = lane >> 4;          // 0: lanes 0-15, 1: lanes 16-31
    const int row0 = blockIdx.x * 128 + wave * 16;
    const int col0 = blockIdx.y * 128;

    // staging assignment: 512 lane-chunks of 16B cover the 8 KB tile;
    // each wave issues 2 async b128 instructions per K-step.
    const int idx0 = (wave * 2 + 0) * 32 + lane;
    const int idx1 = (wave * 2 + 1) * 32 + lane;

    auto stage = [&](int buf, int k0) {
        {
            const int c = idx0 >> 2, q = idx0 & 3;
            const _Float16* g = Wh + (size_t)(col0 + c) * K + k0 + q * 8;
            const unsigned l = (unsigned)(size_t)&Bs[buf][c * BSTRIDE + q * 8];
            asm volatile("global_load_async_to_lds_b128 %0, %1, off"
                         :: "v"(l), "v"(g) : "memory");
        }
        {
            const int c = idx1 >> 2, q = idx1 & 3;
            const _Float16* g = Wh + (size_t)(col0 + c) * K + k0 + q * 8;
            const unsigned l = (unsigned)(size_t)&Bs[buf][c * BSTRIDE + q * 8];
            asm volatile("global_load_async_to_lds_b128 %0, %1, off"
                         :: "v"(l), "v"(g) : "memory");
        }
    };

    v8f acc[8];
#pragma unroll
    for (int t = 0; t < 8; ++t) acc[t] = (v8f){};

    const int  arow  = row0 + r;
    const bool aval  = arow < M;
    const float amask = aval ? 1.f : 0.f;
    const float* Abase = A + (size_t)(aval ? arow : (M - 1)) * K;

    // prologue: stage buffer 0, wait own DMAs, workgroup barrier
    stage(0, 0);
    asm volatile("s_wait_asynccnt 0x0" ::: "memory");
    __syncthreads();

    for (int k0 = 0; k0 < K; k0 += 32) {
        const int buf = (k0 >> 5) & 1;
        if (k0 + 32 < K) stage(buf ^ 1, k0 + 32);

        // A fragment 16x32 f16: lanes 0-15 row=r K={0..7,16..23}; lanes 16-31 K={8..15,24..31}
        const float* Ap = Abase + k0;
        __builtin_prefetch(Ap + 64, 0, 1);
        v16h a;
        const int kb = half * 8;
#pragma unroll
        for (int j = 0; j < 8; ++j) {
            const int kk = kb + (j & 3) * 2 + ((j >> 2) << 4);
            a[2*j]   = (_Float16)(Ap[kk]     * amask);
            a[2*j+1] = (_Float16)(Ap[kk + 1] * amask);
        }
        // preload all 8 B fragments from LDS (distinct regs -> batched ds_loads)
        v16h bfrag[8];
#pragma unroll
        for (int t = 0; t < 8; ++t) {
            const _Float16* bp = &Bs[buf][(t * 16 + r) * BSTRIDE + half * 8];
            const v8h blo = *(const v8h*)bp;
            const v8h bhi = *(const v8h*)(bp + 16);
#pragma unroll
            for (int j = 0; j < 8; ++j) { bfrag[t][j] = blo[j]; bfrag[t][8 + j] = bhi[j]; }
        }
#pragma unroll
        for (int t = 0; t < 8; ++t)
            acc[t] = __builtin_amdgcn_wmma_f32_16x16x32_f16(
                false, a, false, bfrag[t], (short)0, acc[t], false, false);

        asm volatile("s_wait_asynccnt 0x0" ::: "memory");
        __syncthreads();
    }
    // C layout: VGPR v -> row v (lanes 0-15) / v+8 (lanes 16-31), col = lane%16
#pragma unroll
    for (int t = 0; t < 8; ++t) {
        const int col = col0 + t * 16 + r;
        const float badd = bias ? bias[col] : 0.f;
#pragma unroll
        for (int v = 0; v < 8; ++v) {
            const int row = row0 + v + half * 8;
            if (row < M) {
                float x = acc[t][v] + badd;
                if (res) x += res[(size_t)row * N + col];
                C[(size_t)row * N + col] = x;
            }
        }
    }
}

// ---------------------------------------------------------------------------
// LayerNorm over 128 features, one wave per (padded) token; pad rows -> 0.
// ---------------------------------------------------------------------------
__global__ __launch_bounds__(32)
void k_layernorm(const float* __restrict__ h, const float* __restrict__ g,
                 const float* __restrict__ b, float* __restrict__ xnp,
                 int ntok, int pad)
{
    const int rowp = blockIdx.x;
    const int lane = threadIdx.x;
    float* out = xnp + (size_t)rowp * EDIM;
    if (rowp < pad) {
        for (int i = lane; i < EDIM; i += 32) out[i] = 0.f;
        return;
    }
    const float* x = h + (size_t)(rowp - pad) * EDIM;
    float v[4], s = 0.f, s2 = 0.f;
#pragma unroll
    for (int i = 0; i < 4; ++i) {
        v[i] = x[lane + 32 * i];
        s += v[i]; s2 += v[i] * v[i];
    }
    for (int off = 16; off; off >>= 1) {
        s  += __shfl_xor(s,  off, 32);
        s2 += __shfl_xor(s2, off, 32);
    }
    const float mu  = s * (1.f / 128.f);
    const float var = s2 * (1.f / 128.f) - mu * mu;
    const float inv = rsqrtf(var + 1e-5f);
#pragma unroll
    for (int i = 0; i < 4; ++i) {
        const int c = lane + 32 * i;
        out[c] = (v[i] - mu) * inv * g[c] + b[c];
    }
}

// ---------------------------------------------------------------------------
// Landmark means: q_l (with SCALE folded), k_l.  n = m*lseg + i_l (contiguous chunks).
// ---------------------------------------------------------------------------
__global__ __launch_bounds__(32)
void k_landmarks(const float* __restrict__ qkv, float* __restrict__ ql,
                 float* __restrict__ kl, int lseg)
{
    const int h = blockIdx.x >> 6;
    const int m = blockIdx.x & 63;
    const int lane = threadIdx.x;
    const int d = lane & 15;
    const int hp = lane >> 4;
    float sq = 0.f, sk = 0.f;
    for (int i = hp; i < lseg; i += 2) {
        const float* row = qkv + ((size_t)m * lseg + i) * QKVN + h * DHEAD;
        sq += row[d];
        sk += row[EDIM + d];
    }
    sq += __shfl_xor(sq, 16, 32);
    sk += __shfl_xor(sk, 16, 32);
    if (hp == 0) {
        const float inv = 1.f / (float)lseg;
        ql[((size_t)h * LMK + m) * DHEAD + d] = sq * inv * SCALEF;
        kl[((size_t)h * LMK + m) * DHEAD + d] = sk * inv;
    }
}

// ---------------------------------------------------------------------------
// 64x64 matmul in LDS, register writeback (dest may alias sources). 256 threads.
// ---------------------------------------------------------------------------
__device__ __forceinline__
void mm64(float (*A)[LMK], float (*B)[LMK], float (*C)[LMK], int t, float scale)
{
    float c[16];
    __syncthreads();
#pragma unroll
    for (int u = 0; u < 16; ++u) {
        const int e = t + u * 256;
        const int i = e >> 6, j = e & 63;
        float s = 0.f;
#pragma unroll 8
        for (int k = 0; k < LMK; ++k) s += A[i][k] * B[k][j];
        c[u] = s * scale;
    }
    __syncthreads();
#pragma unroll
    for (int u = 0; u < 16; ++u) {
        const int e = t + u * 256;
        C[e >> 6][e & 63] = c[u];
    }
    __syncthreads();
}

// ---------------------------------------------------------------------------
// sim2 -> softmax -> Moore-Penrose pinv (6 iters). One block (256 thr) per head.
// ---------------------------------------------------------------------------
__global__ __launch_bounds__(256)
void k_pinv(const float* __restrict__ ql, const float* __restrict__ kl,
            float* __restrict__ Z)
{
    __shared__ float X[LMK][LMK], Zm[LMK][LMK], T1[LMK][LMK], T2[LMK][LMK];
    const int h = blockIdx.x;
    const int t = threadIdx.x;
    const float* Q = ql + (size_t)h * LMK * DHEAD;
    const float* K = kl + (size_t)h * LMK * DHEAD;

    for (int e = t; e < LMK * LMK; e += 256) {
        const int i = e >> 6, j = e & 63;
        float s = 0.f;
#pragma unroll
        for (int d = 0; d < DHEAD; ++d) s += Q[i * DHEAD + d] * K[j * DHEAD + d];
        X[i][j] = s;
    }
    __syncthreads();
    if (t < LMK) {  // row softmax
        float mx = -3.4e38f;
        for (int j = 0; j < LMK; ++j) mx = fmaxf(mx, X[t][j]);
        float sum = 0.f;
        for (int j = 0; j < LMK; ++j) { const float e = __expf(X[t][j] - mx); X[t][j] = e; sum += e; }
        const float inv = 1.f / sum;
        for (int j = 0; j < LMK; ++j) X[t][j] *= inv;
    }
    __syncthreads();
    if (t < LMK) {  // norm scratch in T1 (not yet live)
        float r = 0.f, c = 0.f;
        for (int j = 0; j < LMK; ++j) { r += fabsf(X[t][j]); c += fabsf(X[j][t]); }
        T1[0][t] = r; T1[1][t] = c;
    }
    __syncthreads();
    if (t == 0) {
        float col = 0.f, row = 0.f;
        for (int i = 0; i < LMK; ++i) { col = fmaxf(col, T1[0][i]); row = fmaxf(row, T1[1][i]); }
        T1[2][0] = 1.f / (col * row);
    }
    __syncthreads();
    const float sc = T1[2][0];
    for (int e = t; e < LMK * LMK; e += 256) {
        const int i = e >> 6, j = e & 63;
        Zm[i][j] = X[j][i] * sc;   // z = x^T / (col*row)
    }
    for (int it = 0; it < 6; ++it) {
        mm64(X, Zm, T1, t, 1.f);                                  // xz
        for (int e = t; e < LMK * LMK; e += 256) {
            const int i = e >> 6, j = e & 63;
            T2[i][j] = (i == j ? 7.f : 0.f) - T1[i][j];
        }
        mm64(T1, T2, T2, t, 1.f);                                 // xz@(7I-xz)
        for (int e = t; e < LMK * LMK; e += 256) {
            const int i = e >> 6, j = e & 63;
            T2[i][j] = (i == j ? 15.f : 0.f) - T2[i][j];
        }
        mm64(T1, T2, T2, t, 1.f);                                 // xz@(15I-..)
        for (int e = t; e < LMK * LMK; e += 256) {
            const int i = e >> 6, j = e & 63;
            T2[i][j] = (i == j ? 13.f : 0.f) - T2[i][j];
        }
        mm64(Zm, T2, Zm, t, 0.25f);                               // z = 0.25*z@(13I-..)
    }
    for (int e = t; e < LMK * LMK; e += 256)
        Z[(size_t)h * LMK * LMK + e] = Zm[e >> 6][e & 63];
}

// ---------------------------------------------------------------------------
// av[h][m][:] = softmax_n(q_l[m] . k[n]) @ v   -- single-pass online softmax.
// ---------------------------------------------------------------------------
__global__ __launch_bounds__(256)
void k_a3v(const float* __restrict__ qkv, const float* __restrict__ ql,
           float* __restrict__ av, int npad)
{
    const int h = blockIdx.x >> 6;
    const int m = blockIdx.x & 63;
    const int t = threadIdx.x;
    __shared__ float qm[DHEAD];
    if (t < DHEAD) qm[t] = ql[((size_t)h * LMK + m) * DHEAD + t];
    __syncthreads();

    float mx = -3.4e38f, sum = 0.f, acc[DHEAD];
#pragma unroll
    for (int d = 0; d < DHEAD; ++d) acc[d] = 0.f;

    for (int n = t; n < npad; n += 256) {
        const float* row = qkv + (size_t)n * QKVN + h * DHEAD;
        float s = 0.f;
#pragma unroll
        for (int d = 0; d < DHEAD; ++d) s += qm[d] * row[EDIM + d];      // k
        const float nm = fmaxf(mx, s);
        const float rs = __expf(mx - nm);
        const float e  = __expf(s - nm);
        sum = sum * rs + e;
#pragma unroll
        for (int d = 0; d < DHEAD; ++d) acc[d] = acc[d] * rs + e * row[2 * EDIM + d];  // v
        mx = nm;
    }
    __shared__ float smx[256], ssum[256], sacc[256][DHEAD];
    smx[t] = mx; ssum[t] = sum;
#pragma unroll
    for (int d = 0; d < DHEAD; ++d) sacc[t][d] = acc[d];
    __syncthreads();
    for (int off = 128; off; off >>= 1) {
        if (t < off) {
            const float m2 = fmaxf(smx[t], smx[t + off]);
            const float a1 = __expf(smx[t] - m2), a2 = __expf(smx[t + off] - m2);
            ssum[t] = ssum[t] * a1 + ssum[t + off] * a2;
#pragma unroll
            for (int d = 0; d < DHEAD; ++d)
                sacc[t][d] = sacc[t][d] * a1 + sacc[t + off][d] * a2;
            smx[t] = m2;
        }
        __syncthreads();
    }
    if (t < DHEAD)
        av[((size_t)h * LMK + m) * DHEAD + t] = sacc[0][t] / ssum[0];
}

// M2[h] = Z[h] @ av[h]   (64x64x16 per head)
__global__ __launch_bounds__(256)
void k_zm(const float* __restrict__ Z, const float* __restrict__ av,
          float* __restrict__ M2)
{
    const int h = blockIdx.x, t = threadIdx.x;
    for (int e = t; e < LMK * DHEAD; e += 256) {
        const int i = e >> 4, d = e & 15;
        float s = 0.f;
#pragma unroll 8
        for (int k = 0; k < LMK; ++k)
            s += Z[(size_t)h * LMK * LMK + i * LMK + k] * av[((size_t)h * LMK + k) * DHEAD + d];
        M2[((size_t)h * LMK + i) * DHEAD + d] = s;
    }
}

// ---------------------------------------------------------------------------
// Per-token attention out + depthwise(33) residual over v.
// ---------------------------------------------------------------------------
__device__ __forceinline__
void attn_row(const float* qkv, const float* kl, const float* M2,
              const float* rw, int n, int npad, int t,
              float* q, float (*p)[LMK], float (*red)[16], float* arow)
{
    const int h = t >> 4, d = t & 15;
    q[t] = qkv[(size_t)n * QKVN + t] * SCALEF;
    __syncthreads();
    const float* qh = &q[h * DHEAD];
    float pv[4];
#pragma unroll
    for (int jj = 0; jj < 4; ++jj) {
        const int j = d * 4 + jj;
        const float* kr = kl + ((size_t)h * LMK + j) * DHEAD;
        float s = 0.f;
#pragma unroll
        for (int e = 0; e < DHEAD; ++e) s += qh[e] * kr[e];
        pv[jj] = s;
    }
    red[h][d] = fmaxf(fmaxf(pv[0], pv[1]), fmaxf(pv[2], pv[3]));
    __syncthreads();
    float mx = red[h][0];
#pragma unroll
    for (int e = 1; e < 16; ++e) mx = fmaxf(mx, red[h][e]);
    __syncthreads();
    float ls = 0.f;
#pragma unroll
    for (int jj = 0; jj < 4; ++jj) {
        pv[jj] = __expf(pv[jj] - mx);
        ls += pv[jj];
        p[h][d * 4 + jj] = pv[jj];
    }
    red[h][d] = ls;
    __syncthreads();
    float sum = 0.f;
#pragma unroll
    for (int e = 0; e < 16; ++e) sum += red[h][e];
    float o = 0.f;
#pragma unroll 8
    for (int j = 0; j < LMK; ++j) o += p[h][j] * M2[((size_t)h * LMK + j) * DHEAD + d];
    o /= sum;
    float r = 0.f;
#pragma unroll
    for (int s = 0; s < 33; ++s) {
        const int nn = n + s - 16;
        if (nn >= 0 && nn < npad)
            r += rw[h * 33 + s] * qkv[(size_t)nn * QKVN + 2 * EDIM + h * DHEAD + d];
    }
    arow[t] = o + r;
}

__global__ __launch_bounds__(128)
void k_attn_out(const float* __restrict__ qkv, const float* __restrict__ kl,
                const float* __restrict__ M2, const float* __restrict__ rw,
                float* __restrict__ att, int npad)
{
    __shared__ float q[EDIM], p[HEADS][LMK], red[HEADS][16], arow[EDIM];
    const int n = blockIdx.x, t = threadIdx.x;
    attn_row(qkv, kl, M2, rw, n, npad, t, q, p, red, arow);
    att[(size_t)n * EDIM + t] = arow[t];
}

// Layer-2: only token 0 (padded index `pad`) is needed -> fused attn-out + out-proj.
__global__ __launch_bounds__(128)
void k_final(const float* __restrict__ qkv, const float* __restrict__ kl,
             const float* __restrict__ M2, const float* __restrict__ rw,
             const float* __restrict__ Wout, const float* __restrict__ bout,
             const float* __restrict__ h3row, float* __restrict__ out,
             int npad, int pad)
{
    __shared__ float q[EDIM], p[HEADS][LMK], red[HEADS][16], arow[EDIM];
    const int t = threadIdx.x;
    attn_row(qkv, kl, M2, rw, pad, npad, t, q, p, red, arow);
    __syncthreads();
    float s = bout[t] + h3row[t];
#pragma unroll 8
    for (int k = 0; k < EDIM; ++k) s += arow[k] * Wout[t * EDIM + k];
    out[t] = s;
}

// ---------------------------------------------------------------------------
// PPEG helpers
// ---------------------------------------------------------------------------
__global__ __launch_bounds__(256)
void k_transpose(const float* __restrict__ h2, float* __restrict__ tbuf, int npix)
{
    const size_t idx = (size_t)blockIdx.x * 256 + threadIdx.x;
    if (idx >= (size_t)npix * EDIM) return;
    const int p = (int)(idx >> 7), c = (int)(idx & 127);
    tbuf[(size_t)c * npix + p] = h2[EDIM + idx];   // skip cls row
}

__global__ __launch_bounds__(256)
void k_ppeg(const float* __restrict__ tbuf,
            const float* __restrict__ w7, const float* __restrict__ b7,
            const float* __restrict__ w5, const float* __restrict__ b5,
            const float* __restrict__ w3, const float* __restrict__ b3,
            float* __restrict__ h3, int Hs)
{
    const int c  = blockIdx.x / Hs;
    const int r  = blockIdx.x % Hs;
    const int cc = threadIdx.x;
    __shared__ float W7[49], W5[25], W3[9];
    if (threadIdx.x < 49) W7[threadIdx.x] = w7[c * 49 + threadIdx.x];
    if (threadIdx.x < 25) W5[threadIdx.x] = w5[c * 25 + threadIdx.x];
    if (threadIdx.x < 9)  W3[threadIdx.x] = w3[c * 9 + threadIdx.x];
    __syncthreads();
    if (cc >= Hs) return;
    const float* T = tbuf + (size_t)c * Hs * Hs;
    float s = T[r * Hs + cc] + b7[c] + b5[c] + b3[c];
#pragma unroll
    for (int dr = -3; dr <= 3; ++dr)
#pragma unroll
        for (int dc = -3; dc <= 3; ++dc) {
            const int rr = r + dr, c2 = cc + dc;
            if (rr >= 0 && rr < Hs && c2 >= 0 && c2 < Hs)
                s += W7[(dr + 3) * 7 + dc + 3] * T[rr * Hs + c2];
        }
#pragma unroll
    for (int dr = -2; dr <= 2; ++dr)
#pragma unroll
        for (int dc = -2; dc <= 2; ++dc) {
            const int rr = r + dr, c2 = cc + dc;
            if (rr >= 0 && rr < Hs && c2 >= 0 && c2 < Hs)
                s += W5[(dr + 2) * 5 + dc + 2] * T[rr * Hs + c2];
        }
#pragma unroll
    for (int dr = -1; dr <= 1; ++dr)
#pragma unroll
        for (int dc = -1; dc <= 1; ++dc) {
            const int rr = r + dr, c2 = cc + dc;
            if (rr >= 0 && rr < Hs && c2 >= 0 && c2 < Hs)
                s += W3[(dr + 1) * 3 + dc + 1] * T[rr * Hs + c2];
        }
    h3[(size_t)(1 + r * Hs + cc) * EDIM + c] = s;
}

__global__ void k_copyrow(const float* __restrict__ src, float* __restrict__ dst)
{
    dst[threadIdx.x] = src[threadIdx.x];
}

// ---------------------------------------------------------------------------
extern "C" void kernel_launch(void* const* d_in, const int* in_sizes, int n_in,
                              void* d_out, int out_size, void* d_ws, size_t ws_size,
                              hipStream_t stream)
{
    const float* features = (const float*)d_in[0];
    const float* fc1_w    = (const float*)d_in[1];
    const float* fc1_b    = (const float*)d_in[2];
    const float* cls_tok  = (const float*)d_in[3];
    const float* ln1_g    = (const float*)d_in[4];
    const float* ln1_b    = (const float*)d_in[5];
    const float* qkv1_w   = (const float*)d_in[6];
    const float* out1_w   = (const float*)d_in[7];
    const float* out1_b   = (const float*)d_in[8];
    const float* res1_w   = (const float*)d_in[9];
    const float* ppeg_w7  = (const float*)d_in[10];
    const float* ppeg_b7  = (const float*)d_in[11];
    const float* ppeg_w5  = (const float*)d_in[12];
    const float* ppeg_b5  = (const float*)d_in[13];
    const float* ppeg_w3  = (const float*)d_in[14];
    const float* ppeg_b3  = (const float*)d_in[15];
    const float* ln2_g    = (const float*)d_in[16];
    const float* ln2_b    = (const float*)d_in[17];
    const float* qkv2_w   = (const float*)d_in[18];
    const float* out2_w   = (const float*)d_in[19];
    const float* out2_b   = (const float*)d_in[20];
    const float* res2_w   = (const float*)d_in[21];

    const int FC_IN = 1024;
    const int N0 = in_sizes[0] / FC_IN;              // 40000
    int Hs = 1; while (Hs * Hs < N0) ++Hs;           // 200 (add==0 for this size)
    const int npix = Hs * Hs;
    const int ntok = npix + 1;                       // 40001
    const int npad = ((ntok + LMK - 1) / LMK) * LMK; // 40064
    const int pad  = npad - ntok;                    // 63
    const int lseg = npad / LMK;                     // 626

    // workspace carve-up
    float* ws   = (float*)d_ws;
    const size_t R = (size_t)npad;
    float* bh   = ws;               // h        R*128
    float* xnp  = bh  + R * 128;    // LN in    R*128 (reused as tbuf)
    float* bq   = xnp + R * 128;    // qkv      R*384
    float* batt = bq  + R * 384;    // attn out R*128 (reused as h3)
    float* bh2  = batt + R * 128;   // h2       R*128
    float* ql   = bh2 + R * 128;
    float* kl   = ql + (size_t)HEADS * LMK * DHEAD;
    float* Zb   = kl + (size_t)HEADS * LMK * DHEAD;
    float* av   = Zb + (size_t)HEADS * LMK * LMK;
    float* M2   = av + (size_t)HEADS * LMK * DHEAD;
    // f16 weight copies
    const int szfc1 = EDIM * FC_IN;     // 131072
    const int szqkv = QKVN * EDIM;      // 49152
    const int szout = EDIM * EDIM;      // 16384
    _Float16* whfc1  = (_Float16*)(M2 + (size_t)HEADS * LMK * DHEAD);
    _Float16* whqkv1 = whfc1 + szfc1;
    _Float16* whout1 = whqkv1 + szqkv;
    _Float16* whqkv2 = whout1 + szout;
    _Float16* whout2 = whqkv2 + szqkv;
    const size_t need = (size_t)((char*)(whout2 + szout) - (char*)ws);
    if (ws_size < need) return;
    float* tbuf = xnp;
    float* bh3  = batt;
    float* outp = (float*)d_out;

    const dim3 blk256(256), blk128(128), blk32(32);
    const int gM  = (N0 + 127) / 128;          // fc1 rows (40000)
    const int gMp = (npad + 127) / 128;        // padded rows
    const int gMt = (ntok + 127) / 128;        // token rows

    // ---- weight pre-conversion to f16 ----
    k_f16cvt<<<(szfc1 + 255) / 256, blk256, 0, stream>>>(fc1_w,  whfc1,  szfc1);
    k_f16cvt<<<(szqkv + 255) / 256, blk256, 0, stream>>>(qkv1_w, whqkv1, szqkv);
    k_f16cvt<<<(szout + 255) / 256, blk256, 0, stream>>>(out1_w, whout1, szout);
    k_f16cvt<<<(szqkv + 255) / 256, blk256, 0, stream>>>(qkv2_w, whqkv2, szqkv);
    k_f16cvt<<<(szout + 255) / 256, blk256, 0, stream>>>(out2_w, whout2, szout);

    // ---- layer 1 ----
    k_copyrow<<<1, 128, 0, stream>>>(cls_tok, bh);
    k_gemm<<<dim3(gM, 1), blk256, 0, stream>>>(features, whfc1, fc1_b, nullptr,
                                               bh + EDIM, N0, FC_IN, EDIM);
    k_layernorm<<<npad, blk32, 0, stream>>>(bh, ln1_g, ln1_b, xnp, ntok, pad);
    k_gemm<<<dim3(gMp, QKVN / 128), blk256, 0, stream>>>(xnp, whqkv1, nullptr, nullptr,
                                                         bq, npad, EDIM, QKVN);
    k_landmarks<<<HEADS * LMK, blk32, 0, stream>>>(bq, ql, kl, lseg);
    k_pinv<<<HEADS, blk256, 0, stream>>>(ql, kl, Zb);
    k_a3v<<<HEADS * LMK, blk256, 0, stream>>>(bq, ql, av, npad);
    k_zm<<<HEADS, blk256, 0, stream>>>(Zb, av, M2);
    k_attn_out<<<npad, blk128, 0, stream>>>(bq, kl, M2, res1_w, batt, npad);
    k_gemm<<<dim3(gMt, 1), blk256, 0, stream>>>(batt + (size_t)pad * EDIM, whout1,
                                                out1_b, bh, bh2, ntok, EDIM, EDIM);
    // ---- PPEG ----
    k_transpose<<<(int)(((size_t)npix * EDIM + 255) / 256), blk256, 0, stream>>>(bh2, tbuf, npix);
    k_ppeg<<<EDIM * Hs, blk256, 0, stream>>>(tbuf, ppeg_w7, ppeg_b7, ppeg_w5, ppeg_b5,
                                             ppeg_w3, ppeg_b3, bh3, Hs);
    k_copyrow<<<1, 128, 0, stream>>>(bh2, bh3);
    // ---- layer 2 (only token 0's output is needed downstream) ----
    k_layernorm<<<npad, blk32, 0, stream>>>(bh3, ln2_g, ln2_b, xnp, ntok, pad);
    k_gemm<<<dim3(gMp, QKVN / 128), blk256, 0, stream>>>(xnp, whqkv2, nullptr, nullptr,
                                                         bq, npad, EDIM, QKVN);
    k_landmarks<<<HEADS * LMK, blk32, 0, stream>>>(bq, ql, kl, lseg);
    k_pinv<<<HEADS, blk256, 0, stream>>>(ql, kl, Zb);
    k_a3v<<<HEADS * LMK, blk256, 0, stream>>>(bq, ql, av, npad);
    k_zm<<<HEADS, blk256, 0, stream>>>(Zb, av, M2);
    k_final<<<1, blk128, 0, stream>>>(bq, kl, M2, res2_w, out2_w, out2_b,
                                      bh3, outp, npad, pad);
}
